// DeepseekMoE_65420941852890
// MI455X (gfx1250) — matrix operand order repaired and checked by reading, not compiled, not run on hardware
//
#include <hip/hip_runtime.h>
#include <hip/hip_bf16.h>
#include <math.h>

#define T_TOKENS 4096
#define NEXP 64
#define TOPK 6
#define NGROUP 8
#define TOPKG 3
#define HID 2048
#define INTER 1408
#define BM 64
#define BN 64
#define BK 32
#define LSTR 20                                        // LDS row stride (dwords): 16-B aligned rows
#define MAXROWS (T_TOKENS * TOPK + NEXP * BM)          // 28672
#define MAXTILES (T_TOKENS * TOPK / BM + NEXP)         // 448

typedef __attribute__((ext_vector_type(16))) __bf16 v16bf;
typedef __attribute__((ext_vector_type(2)))  __bf16 v2bf;
typedef __attribute__((ext_vector_type(2)))  float  v2f;
typedef __attribute__((ext_vector_type(8)))  float  v8f;

union FragU { unsigned int u[8]; uint4 q[2]; v16bf v; };

#if defined(__has_builtin)
#if __has_builtin(__builtin_amdgcn_global_load_async_to_lds_b128)
#define USE_ASYNC_LDS 1
#endif
#endif

#ifdef USE_ASYNC_LDS
typedef int async_v4i __attribute__((vector_size(16)));
#endif

// f32x2 -> bf16x2 as a single vector conversion: lowers to one v_cvt_pk_bf16_f32.
__device__ __forceinline__ unsigned int pack_bf16x2(float a, float b) {
  v2f f;
  f.x = a;
  f.y = b;
  v2bf h = __builtin_convertvector(f, v2bf);
  union { v2bf v; unsigned int u; } r;
  r.v = h;
  return r.u;
}
__device__ __forceinline__ unsigned short f2bf(float a) {
  union { __bf16 h; unsigned short u; } r;
  r.h = (__bf16)a;
  return r.u;
}
__device__ __forceinline__ void wait_async_lds() {
#ifdef USE_ASYNC_LDS
#if __has_builtin(__builtin_amdgcn_s_wait_asynccnt)
  __builtin_amdgcn_s_wait_asynccnt(0);
#else
  asm volatile("s_wait_asynccnt 0x0" ::: "memory");
#endif
#endif
}
// async (or sync fallback) 16-byte global->LDS copy
__device__ __forceinline__ void copy16_to_lds(const void* gsrc, unsigned int* ldst) {
#ifdef USE_ASYNC_LDS
  async_v4i* gp = (async_v4i*)const_cast<void*>(gsrc);
  __attribute__((address_space(1))) async_v4i* gp1 =
      (__attribute__((address_space(1))) async_v4i*)gp;
  async_v4i* lp = (async_v4i*)ldst;
  __attribute__((address_space(3))) async_v4i* lp3 =
      (__attribute__((address_space(3))) async_v4i*)lp;
  __builtin_amdgcn_global_load_async_to_lds_b128(gp1, lp3, 0, 0);
#else
  *(uint4*)ldst = *(const uint4*)gsrc;
#endif
}

// ---------------- init / zero / x-conversion ----------------
__global__ void init_kernel(int* n_tiles, int* expert_count, int* cursor) {
  int tid = threadIdx.x;
  if (tid == 0) n_tiles[0] = 0;
  if (tid < NEXP) { expert_count[tid] = 0; cursor[tid] = 0; }
}

__global__ void zero_out_kernel(float4* out, int n4) {
  int i = blockIdx.x * 256 + threadIdx.x;
  if (i < n4) out[i] = make_float4(0.f, 0.f, 0.f, 0.f);
}

// hidden_states f32 -> bf16 (one pass; removes per-N-tile reconversion in gemm1)
__global__ void convx_kernel(const float4* __restrict__ x4, uint4* __restrict__ xb4, int n8) {
  int i = blockIdx.x * 256 + threadIdx.x;
  if (i >= n8) return;
  float4 f0 = x4[2 * i], f1 = x4[2 * i + 1];
  uint4 o;
  o.x = pack_bf16x2(f0.x, f0.y);
  o.y = pack_bf16x2(f0.z, f0.w);
  o.z = pack_bf16x2(f1.x, f1.y);
  o.w = pack_bf16x2(f1.z, f1.w);
  xb4[i] = o;
}

// ---------------- router: logits + softmax + grouped top-k ----------------
__global__ __launch_bounds__(256) void router_kernel(
    const float* __restrict__ x, const float* __restrict__ gw,
    int* __restrict__ topk_id, float* __restrict__ topk_w,
    int* __restrict__ expert_count) {
  __shared__ float part[256];
  __shared__ float logits[NEXP];
  const int t = blockIdx.x, tid = threadIdx.x;
  const int e = tid >> 2, p = tid & 3;
  const float4* x4 = (const float4*)(x + (size_t)t * HID + p * (HID / 4));
  const float4* w4 = (const float4*)(gw + (size_t)e * HID + p * (HID / 4));
  float s = 0.f;
  for (int k = 0; k < HID / 16; ++k) {
    float4 a = x4[k], b = w4[k];
    s += a.x * b.x + a.y * b.y + a.z * b.z + a.w * b.w;
  }
  part[tid] = s;
  __syncthreads();
  if (tid < NEXP)
    logits[tid] = part[tid * 4] + part[tid * 4 + 1] + part[tid * 4 + 2] + part[tid * 4 + 3];
  __syncthreads();
  if (tid == 0) {
    float sc[NEXP];
    float mx = -1e30f;
    for (int i = 0; i < NEXP; ++i) mx = fmaxf(mx, logits[i]);
    float sum = 0.f;
    for (int i = 0; i < NEXP; ++i) { sc[i] = __expf(logits[i] - mx); sum += sc[i]; }
    float inv = 1.f / sum;
    for (int i = 0; i < NEXP; ++i) sc[i] *= inv;
    float gs[NGROUP];
    for (int g = 0; g < NGROUP; ++g) {
      float m = sc[g * 8];
      for (int j = 1; j < 8; ++j) m = fmaxf(m, sc[g * 8 + j]);
      gs[g] = m;
    }
    bool gsel[NGROUP];
    for (int g = 0; g < NGROUP; ++g) gsel[g] = false;
    for (int r = 0; r < TOPKG; ++r) {
      int best = 0; float bv = -1e30f;
      for (int g = 0; g < NGROUP; ++g)
        if (!gsel[g] && gs[g] > bv) { bv = gs[g]; best = g; }
      gsel[best] = true;
    }
    float masked[NEXP];
    for (int i = 0; i < NEXP; ++i) masked[i] = gsel[i >> 3] ? sc[i] : 0.f;
    int ids[TOPK]; float ws[TOPK]; float wsum = 0.f;
    for (int r = 0; r < TOPK; ++r) {
      int best = 0; float bv = -1.f;
      for (int i = 0; i < NEXP; ++i)
        if (masked[i] > bv) { bv = masked[i]; best = i; }
      ids[r] = best; ws[r] = bv; masked[best] = -1.f; wsum += bv;
    }
    float wi = 1.f / (wsum + 1e-20f);
    for (int r = 0; r < TOPK; ++r) {
      topk_id[t * TOPK + r] = ids[r];
      topk_w[t * TOPK + r] = ws[r] * wi;
      atomicAdd(&expert_count[ids[r]], 1);
    }
  }
}

// ---------------- plan: padded offsets + tile list ----------------
__global__ void plan_kernel(const int* __restrict__ expert_count,
                            int* __restrict__ pad_off, int* __restrict__ tile_e,
                            int* __restrict__ tile_r, int* __restrict__ n_tiles) {
  if (threadIdx.x == 0 && blockIdx.x == 0) {
    int off = 0, nt = 0;
    for (int e = 0; e < NEXP; ++e) {
      pad_off[e] = off;
      int c = expert_count[e];
      int tiles = (c + BM - 1) / BM;
      for (int i = 0; i < tiles; ++i) { tile_e[nt] = e; tile_r[nt] = off + i * BM; ++nt; }
      off += tiles * BM;
    }
    pad_off[NEXP] = off;
    n_tiles[0] = nt;
  }
}

// ---------------- scatter tokens into expert-contiguous rows ----------------
__global__ void scatter_kernel(const int* __restrict__ topk_id,
                               const float* __restrict__ topk_w,
                               const int* __restrict__ pad_off, int* __restrict__ cursor,
                               int* __restrict__ token_list, float* __restrict__ row_w) {
  int i = blockIdx.x * 256 + threadIdx.x;
  if (i >= T_TOKENS * TOPK) return;
  int e = topk_id[i];
  int pos = pad_off[e] + atomicAdd(&cursor[e], 1);
  token_list[pos] = i / TOPK;
  row_w[pos] = topk_w[i];
}

__global__ void pad_kernel(const int* __restrict__ expert_count,
                           const int* __restrict__ pad_off,
                           int* __restrict__ token_list, float* __restrict__ row_w) {
  int e = blockIdx.x;
  int c = expert_count[e];
  int base = pad_off[e];
  int padded = pad_off[e + 1] - base;
  int fill = (c > 0) ? token_list[base] : 0;
  for (int r = c + threadIdx.x; r < padded; r += blockDim.x) {
    token_list[base + r] = fill;
    row_w[base + r] = 0.f;
  }
}

// ---------------- grouped GEMM 1: act = silu(X Wg) * (X Wu), bf16 WMMA ----------------
__global__ __launch_bounds__(256) void gemm1_kernel(
    const unsigned short* __restrict__ xb, const float* __restrict__ w_gate,
    const float* __restrict__ w_up, const int* __restrict__ n_tiles,
    const int* __restrict__ tile_e, const int* __restrict__ tile_r,
    const int* __restrict__ token_list, unsigned short* __restrict__ act) {
  const int tile = blockIdx.y;
  if (tile >= n_tiles[0]) return;
  const int e = tile_e[tile];
  const int row0 = tile_r[tile];
  const int n0 = blockIdx.x * BN;
  const float* Wg = w_gate + (size_t)e * HID * INTER;
  const float* Wu = w_up + (size_t)e * HID * INTER;

  __shared__ __align__(16) unsigned int Asu[BM][LSTR];  // bf16 K-pairs, row = M
  __shared__ __align__(16) unsigned int Bgf[BN][LSTR];  // fragment-major: row = N, col = K-pair
  __shared__ __align__(16) unsigned int Buf[BN][LSTR];

  const int tid = threadIdx.x;
  const int lane = tid & 31, wid = tid >> 5;
  const int wr = wid >> 1, wc = wid & 1;            // wave: rows [wr*16,+16), cols [wc*32,+32)
  const int arow = tid >> 2;                        // 0..63
  const int akq = (tid & 3) * 8;                    // 0,8,16,24
  const int bn = tid & 63;                          // 0..63
  const int bkp0 = tid >> 6;                        // 0..3

  // loop-invariant global row pointers (token gather resolved once); xb is bf16
  const unsigned short* xrowb = xb + (size_t)token_list[row0 + arow] * HID + akq;
  const float* wgp = Wg + (size_t)(2 * bkp0) * INTER + n0 + bn;
  const float* wup = Wu + (size_t)(2 * bkp0) * INTER + n0 + bn;

  v8f cg0 = {}, cg1 = {}, cu0 = {}, cu1 = {};

  // stage B tile kk=0 into registers
  float sg0[4], sg1[4], su0[4], su1[4];
#pragma unroll
  for (int l = 0; l < 4; ++l) {
    size_t o = (size_t)(8 * l) * INTER;
    sg0[l] = wgp[o]; sg1[l] = wgp[o + INTER];
    su0[l] = wup[o]; su1[l] = wup[o + INTER];
  }

  for (int kk = 0; kk < HID; kk += BK) {
    // A tile: pure bf16 copy -> async global->LDS (ASYNCcnt)
    copy16_to_lds(xrowb + kk, &Asu[arow][akq / 2]);
    // staged B registers -> LDS (single packed cvt per pair)
#pragma unroll
    for (int l = 0; l < 4; ++l) {
      Bgf[bn][bkp0 + 4 * l] = pack_bf16x2(sg0[l], sg1[l]);
      Buf[bn][bkp0 + 4 * l] = pack_bf16x2(su0[l], su1[l]);
    }
    // issue next tile's B loads; latency overlaps the WMMAs below
    if (kk + BK < HID) {
      const float* wg2 = wgp + (size_t)(kk + BK) * INTER;
      const float* wu2 = wup + (size_t)(kk + BK) * INTER;
#pragma unroll
      for (int l = 0; l < 4; ++l) {
        size_t o = (size_t)(8 * l) * INTER;
        sg0[l] = wg2[o]; sg1[l] = wg2[o + INTER];
        su0[l] = wu2[o]; su1[l] = wu2[o + INTER];
      }
    }
    wait_async_lds();
    __syncthreads();
    // fragments: contiguous 16-B LDS reads (ds_load_b128)
    const int h = lane >> 4, ml = lane & 15;
    FragU a, bg0, bg1, bu0, bu1;
    const unsigned int* arp = &Asu[wr * 16 + ml][0];
    a.q[0] = *(const uint4*)(arp + h * 8);
    a.q[1] = *(const uint4*)(arp + h * 8 + 4);
    const int nb0 = wc * 32 + ml, nb1 = nb0 + 16;
    bg0.q[0] = *(const uint4*)&Bgf[nb0][h * 8];
    bg0.q[1] = *(const uint4*)&Bgf[nb0][h * 8 + 4];
    bg1.q[0] = *(const uint4*)&Bgf[nb1][h * 8];
    bg1.q[1] = *(const uint4*)&Bgf[nb1][h * 8 + 4];
    bu0.q[0] = *(const uint4*)&Buf[nb0][h * 8];
    bu0.q[1] = *(const uint4*)&Buf[nb0][h * 8 + 4];
    bu1.q[0] = *(const uint4*)&Buf[nb1][h * 8];
    bu1.q[1] = *(const uint4*)&Buf[nb1][h * 8 + 4];
    cg0 = __builtin_amdgcn_wmma_f32_16x16x32_bf16(false, a.v, false, bg0.v, (short)0, cg0, false, false);
    cg1 = __builtin_amdgcn_wmma_f32_16x16x32_bf16(false, a.v, false, bg1.v, (short)0, cg1, false, false);
    cu0 = __builtin_amdgcn_wmma_f32_16x16x32_bf16(false, a.v, false, bu0.v, (short)0, cu0, false, false);
    cu1 = __builtin_amdgcn_wmma_f32_16x16x32_bf16(false, a.v, false, bu1.v, (short)0, cu1, false, false);
    __syncthreads();
  }
  // fused SiLU(gate) * up, store bf16 activations
  const int h = lane >> 4, nl = lane & 15;
#pragma unroll
  for (int i = 0; i < 8; ++i) {
    int m = h * 8 + i;
    size_t rbase = (size_t)(row0 + wr * 16 + m) * INTER + n0 + wc * 32;
    float g0 = cg0[i], a0 = (g0 / (1.f + __expf(-g0))) * cu0[i];
    float g1 = cg1[i], a1 = (g1 / (1.f + __expf(-g1))) * cu1[i];
    act[rbase + nl] = f2bf(a0);
    act[rbase + 16 + nl] = f2bf(a1);
  }
}

// ---------------- grouped GEMM 2: out += w * (act Wd) ----------------
__global__ __launch_bounds__(256) void gemm2_kernel(
    const unsigned short* __restrict__ act, const float* __restrict__ w_down,
    const int* __restrict__ n_tiles, const int* __restrict__ tile_e,
    const int* __restrict__ tile_r, const int* __restrict__ token_list,
    const float* __restrict__ row_w, float* __restrict__ out) {
  const int tile = blockIdx.y;
  if (tile >= n_tiles[0]) return;
  const int e = tile_e[tile];
  const int row0 = tile_r[tile];
  const int n0 = blockIdx.x * BN;
  const float* Wd = w_down + (size_t)e * INTER * HID;

  __shared__ __align__(16) unsigned int Asu[BM][LSTR];
  __shared__ __align__(16) unsigned int Bf[BN][LSTR];
  __shared__ int toks[BM];
  __shared__ float rws[BM];

  const int tid = threadIdx.x;
  if (tid < BM) { toks[tid] = token_list[row0 + tid]; rws[tid] = row_w[row0 + tid]; }

  const int lane = tid & 31, wid = tid >> 5;
  const int wr = wid >> 1, wc = wid & 1;
  const int arow = tid >> 2, akq = (tid & 3) * 8;
  const int bn = tid & 63, bkp0 = tid >> 6;

  const unsigned short* arowp = act + (size_t)(row0 + arow) * INTER + akq;
  const float* wdp = Wd + (size_t)(2 * bkp0) * HID + n0 + bn;

  v8f c0 = {}, c1 = {};

  float sd0[4], sd1[4];
#pragma unroll
  for (int l = 0; l < 4; ++l) {
    size_t o = (size_t)(8 * l) * HID;
    sd0[l] = wdp[o]; sd1[l] = wdp[o + HID];
  }

  for (int kk = 0; kk < INTER; kk += BK) {
    // A tile: pure bf16 copy -> async global->LDS (ASYNCcnt)
    copy16_to_lds(arowp + kk, &Asu[arow][akq / 2]);
    // staged B registers -> LDS
#pragma unroll
    for (int l = 0; l < 4; ++l) {
      Bf[bn][bkp0 + 4 * l] = pack_bf16x2(sd0[l], sd1[l]);
    }
    // issue next tile's B loads before waiting
    if (kk + BK < INTER) {
      const float* wd2 = wdp + (size_t)(kk + BK) * HID;
#pragma unroll
      for (int l = 0; l < 4; ++l) {
        size_t o = (size_t)(8 * l) * HID;
        sd0[l] = wd2[o]; sd1[l] = wd2[o + HID];
      }
    }
    wait_async_lds();
    __syncthreads();
    const int h = lane >> 4, ml = lane & 15;
    FragU a, b0, b1;
    const unsigned int* arp = &Asu[wr * 16 + ml][0];
    a.q[0] = *(const uint4*)(arp + h * 8);
    a.q[1] = *(const uint4*)(arp + h * 8 + 4);
    const int nb0 = wc * 32 + ml, nb1 = nb0 + 16;
    b0.q[0] = *(const uint4*)&Bf[nb0][h * 8];
    b0.q[1] = *(const uint4*)&Bf[nb0][h * 8 + 4];
    b1.q[0] = *(const uint4*)&Bf[nb1][h * 8];
    b1.q[1] = *(const uint4*)&Bf[nb1][h * 8 + 4];
    c0 = __builtin_amdgcn_wmma_f32_16x16x32_bf16(false, a.v, false, b0.v, (short)0, c0, false, false);
    c1 = __builtin_amdgcn_wmma_f32_16x16x32_bf16(false, a.v, false, b1.v, (short)0, c1, false, false);
    __syncthreads();
  }
  const int h = lane >> 4, nl = lane & 15;
#pragma unroll
  for (int i = 0; i < 8; ++i) {
    int m = wr * 16 + h * 8 + i;
    float w = rws[m];
    int tkn = toks[m];
    float* obase = out + (size_t)tkn * HID + n0 + wc * 32;
    unsafeAtomicAdd(obase + nl, c0[i] * w);
    unsafeAtomicAdd(obase + 16 + nl, c1[i] * w);
  }
}

// ---------------- host launch ----------------
extern "C" void kernel_launch(void* const* d_in, const int* in_sizes, int n_in,
                              void* d_out, int out_size, void* d_ws, size_t ws_size,
                              hipStream_t stream) {
  const float* x  = (const float*)d_in[0];   // [T, H]
  const float* gw = (const float*)d_in[1];   // [E, H]
  const float* wg = (const float*)d_in[2];   // [E, H, I]
  const float* wu = (const float*)d_in[3];   // [E, H, I]
  const float* wd = (const float*)d_in[4];   // [E, I, H]
  float* out = (float*)d_out;                // [T, H]

  char* p = (char*)d_ws;
  auto carve = [&](size_t bytes) -> char* {
    char* r = p;
    p += (bytes + 255) & ~(size_t)255;
    return r;
  };
  int* n_tiles       = (int*)carve(sizeof(int));
  int* expert_count  = (int*)carve(NEXP * sizeof(int));
  int* pad_off       = (int*)carve((NEXP + 1) * sizeof(int));
  int* cursor        = (int*)carve(NEXP * sizeof(int));
  int* tile_e        = (int*)carve(MAXTILES * sizeof(int));
  int* tile_r        = (int*)carve(MAXTILES * sizeof(int));
  int* topk_id       = (int*)carve((size_t)T_TOKENS * TOPK * sizeof(int));
  float* topk_w      = (float*)carve((size_t)T_TOKENS * TOPK * sizeof(float));
  int* token_list    = (int*)carve(MAXROWS * sizeof(int));
  float* row_w       = (float*)carve(MAXROWS * sizeof(float));
  unsigned short* xb  = (unsigned short*)carve((size_t)T_TOKENS * HID * sizeof(unsigned short));
  unsigned short* act = (unsigned short*)carve((size_t)MAXROWS * INTER * sizeof(unsigned short));

  init_kernel<<<1, 256, 0, stream>>>(n_tiles, expert_count, cursor);
  const int nOut4 = T_TOKENS * HID / 4;
  zero_out_kernel<<<(nOut4 + 255) / 256, 256, 0, stream>>>((float4*)out, nOut4);
  const int n8 = T_TOKENS * HID / 8;
  convx_kernel<<<(n8 + 255) / 256, 256, 0, stream>>>((const float4*)x, (uint4*)xb, n8);
  router_kernel<<<T_TOKENS, 256, 0, stream>>>(x, gw, topk_id, topk_w, expert_count);
  plan_kernel<<<1, 1, 0, stream>>>(expert_count, pad_off, tile_e, tile_r, n_tiles);
  scatter_kernel<<<(T_TOKENS * TOPK + 255) / 256, 256, 0, stream>>>(
      topk_id, topk_w, pad_off, cursor, token_list, row_w);
  pad_kernel<<<NEXP, 64, 0, stream>>>(expert_count, pad_off, token_list, row_w);
  gemm1_kernel<<<dim3(INTER / BN, MAXTILES), 256, 0, stream>>>(
      xb, wg, wu, n_tiles, tile_e, tile_r, token_list, act);
  gemm2_kernel<<<dim3(HID / BN, MAXTILES), 256, 0, stream>>>(
      act, wd, n_tiles, tile_e, tile_r, token_list, row_w, out);
}
